// SParameterPredictionModel_67611375174106
// MI455X (gfx1250) — matrix operand order repaired and checked by reading, not compiled, not run on hardware
//
#include <hip/hip_runtime.h>

// Model dims
#define BB 8
#define TT 1024
#define INDIM 16
#define DD 256
#define HH 8
#define LLAYERS 4
#define DKK 32
#define MTOT (BB * TT) // 8192

typedef __attribute__((ext_vector_type(16))) __bf16 v16bf;
typedef __attribute__((ext_vector_type(8)))  __bf16 v8bf;
typedef __attribute__((ext_vector_type(8)))  float  v8f;

static __device__ __forceinline__ v8f wmma_bf16(v16bf a, v16bf b, v8f c) {
  // (neg_a, A, neg_b, B, c_mod, C, reuse_a, reuse_b)
  return __builtin_amdgcn_wmma_f32_16x16x32_bf16(false, a, false, b, (short)0, c, false, false);
}

// Load a 16x32 bf16 A-tile (row-major, leading dim `ld`) into the ISA A-layout:
// lanes 0-15 (M=lane): elems 0..7 = K 0..7, elems 8..15 = K 16..23
// lanes 16-31 (M=lane-16): elems 0..7 = K 8..15, elems 8..15 = K 24..31
static __device__ __forceinline__ v16bf load_a16x32(const __bf16* tile, int ld, int lane) {
  const int m = lane & 15;
  const int koff = (lane >> 4) << 3; // 0 or 8
  const __bf16* p = tile + (size_t)m * ld + koff;
  v8bf lo = *reinterpret_cast<const v8bf*>(p);
  v8bf hi = *reinterpret_cast<const v8bf*>(p + 16);
  return __builtin_shufflevector(lo, hi, 0, 1, 2, 3, 4, 5, 6, 7, 8, 9, 10, 11, 12, 13, 14, 15);
}

// ---------------- fp32 -> bf16 weight conversion ----------------
__global__ void cvt_bf16_kernel(const float* __restrict__ src, __bf16* __restrict__ dst, int n) {
  int i = blockIdx.x * blockDim.x + threadIdx.x;
  if (i < n) dst[i] = (__bf16)src[i];
}

// ---------------- input projection: h = x @ Wp + bp + pos ----------------
__global__ void inproj_kernel(const float* __restrict__ x, const float* __restrict__ Wp,
                              const float* __restrict__ bp, const float* __restrict__ pos,
                              float* __restrict__ h, __bf16* __restrict__ hb) {
  int row = blockIdx.x;        // b*T + t
  int n = threadIdx.x;         // 0..255
  int t = row & (TT - 1);
  float acc = bp[n] + pos[(size_t)t * DD + n];
  const float* xr = x + (size_t)row * INDIM;
#pragma unroll
  for (int k = 0; k < INDIM; ++k) acc += xr[k] * Wp[(size_t)k * DD + n];
  h[(size_t)row * DD + n] = acc;
  hb[(size_t)row * DD + n] = (__bf16)acc;
}

// ---------------- QKV GEMM: [8192x256] @ [256x768] + bias, scattered to Q/Kt/V ----------------
__global__ __launch_bounds__(32) void qkv_gemm_kernel(
    const __bf16* __restrict__ A, const __bf16* __restrict__ W, const float* __restrict__ bias,
    __bf16* __restrict__ Qb, __bf16* __restrict__ Ktb, __bf16* __restrict__ Vb) {
  const int lane = threadIdx.x;
  const int m0 = blockIdx.x * 16;
  const int n0 = blockIdx.y * 64;
  v8f acc0 = {}, acc1 = {}, acc2 = {}, acc3 = {};
  for (int kk = 0; kk < DD; kk += 32) {
    v16bf a = load_a16x32(A + (size_t)m0 * DD + kk, DD, lane);
    // B-layout: lane L holds column N = n0+(L&15) (+16j), 16 K values in table order
    v16bf b0, b1, b2, b3;
#pragma unroll
    for (int s = 0; s < 16; ++s) {
      const int k = kk + ((lane >> 4) << 3) + (s & 7) + ((s >> 3) << 4);
      const __bf16* wp = W + (size_t)k * (3 * DD) + n0 + (lane & 15);
      b0[s] = wp[0]; b1[s] = wp[16]; b2[s] = wp[32]; b3[s] = wp[48];
    }
    acc0 = wmma_bf16(a, b0, acc0);
    acc1 = wmma_bf16(a, b1, acc1);
    acc2 = wmma_bf16(a, b2, acc2);
    acc3 = wmma_bf16(a, b3, acc3);
  }
  const int mbase = (lane >> 4) << 3;
  const int nlo = lane & 15;
  v8f accs[4] = {acc0, acc1, acc2, acc3};
#pragma unroll
  for (int j = 0; j < 4; ++j) {
    int n = n0 + j * 16 + nlo;
    int mat = n >> 8;            // 0=Q, 1=K, 2=V (uniform across the wave)
    int hd = (n & 255) >> 5;
    int dk = n & 31;
    float bn = bias[n];
#pragma unroll
    for (int r = 0; r < 8; ++r) {
      int row = m0 + mbase + r;
      int b_ = row >> 10;
      int t = row & 1023;
      __bf16 bv = (__bf16)(accs[j][r] + bn);
      int bh = b_ * HH + hd;
      if (mat == 0)      Qb[((size_t)bh * TT + t) * DKK + dk] = bv;       // [b][h][t][dk]
      else if (mat == 1) Ktb[((size_t)bh * TT + t) * DKK + dk] = bv;      // [b][h][t][dk]
      else               Vb[((size_t)bh * DKK + dk) * TT + t] = bv;       // [b][h][dk][t] (transposed)
    }
  }
}

// ---------------- attention: one wave per (b,h, 16-query tile); scores in LDS ----------------
__global__ __launch_bounds__(32) void attn_kernel(
    const __bf16* __restrict__ Qb, const __bf16* __restrict__ Ktb,
    const __bf16* __restrict__ Vb, __bf16* __restrict__ Ob) {
  __shared__ float S[16 * TT]; // 64 KB of the 320 KB WGP LDS
  const int lane = threadIdx.x;
  const int bh = blockIdx.x >> 6;   // b*H + h
  const int qt = blockIdx.x & 63;   // query tile
  const int b_ = bh >> 3;
  const int hd = bh & 7;
  const float scale = 0.17677669529663688f; // 1/sqrt(32)

  v16bf aQ = load_a16x32(Qb + ((size_t)bh * TT + qt * 16) * DKK, DKK, lane);

  const int mtop = (lane >> 4) << 3;
  const int nlo = lane & 15;

  // Phase 1: S = scale * Q @ K^T  (K is [t][dk] so B-operand loads like an A-tile: lane = key, dk in table order)
  for (int kc = 0; kc < TT / 16; ++kc) {
    v16bf bK = load_a16x32(Ktb + ((size_t)bh * TT + kc * 16) * DKK, DKK, lane);
    v8f z = {};
    v8f s = wmma_bf16(aQ, bK, z);
#pragma unroll
    for (int r = 0; r < 8; ++r)
      S[(mtop + r) * TT + kc * 16 + nlo] = s[r] * scale;
  }

  // Phase 2: per-row max and sum(exp); lane keeps stats for row (lane&15)
  float my_max = 0.f, my_isum = 0.f;
#pragma unroll 1
  for (int m = 0; m < 16; ++m) {
    const float4* row4 = reinterpret_cast<const float4*>(&S[m * TT]);
    float mx = -3.0e38f;
    for (int c = lane; c < TT / 4; c += 32) {
      float4 v = row4[c];
      mx = fmaxf(mx, fmaxf(fmaxf(v.x, v.y), fmaxf(v.z, v.w)));
    }
#pragma unroll
    for (int off = 16; off > 0; off >>= 1) mx = fmaxf(mx, __shfl_xor(mx, off));
    float sm = 0.f;
    for (int c = lane; c < TT / 4; c += 32) {
      float4 v = row4[c];
      sm += __expf(v.x - mx) + __expf(v.y - mx) + __expf(v.z - mx) + __expf(v.w - mx);
    }
#pragma unroll
    for (int off = 16; off > 0; off >>= 1) sm += __shfl_xor(sm, off);
    if ((lane & 15) == m) { my_max = mx; my_isum = 1.0f / sm; }
  }

  // Phase 3: O = softmax(S) @ V, rebuilding P directly in A-layout from LDS
  v8f o0 = {}, o1 = {};
  const int mA = lane & 15;
  const int koff = (lane >> 4) << 3;
  for (int kc = 0; kc < TT / 32; ++kc) {
    const float* srow = &S[mA * TT + kc * 32];
    float4 sa = *reinterpret_cast<const float4*>(srow + koff);
    float4 sb = *reinterpret_cast<const float4*>(srow + koff + 4);
    float4 sc = *reinterpret_cast<const float4*>(srow + 16 + koff);
    float4 sd = *reinterpret_cast<const float4*>(srow + 16 + koff + 4);
    v16bf aP;
    aP[0]  = (__bf16)(__expf(sa.x - my_max) * my_isum);
    aP[1]  = (__bf16)(__expf(sa.y - my_max) * my_isum);
    aP[2]  = (__bf16)(__expf(sa.z - my_max) * my_isum);
    aP[3]  = (__bf16)(__expf(sa.w - my_max) * my_isum);
    aP[4]  = (__bf16)(__expf(sb.x - my_max) * my_isum);
    aP[5]  = (__bf16)(__expf(sb.y - my_max) * my_isum);
    aP[6]  = (__bf16)(__expf(sb.z - my_max) * my_isum);
    aP[7]  = (__bf16)(__expf(sb.w - my_max) * my_isum);
    aP[8]  = (__bf16)(__expf(sc.x - my_max) * my_isum);
    aP[9]  = (__bf16)(__expf(sc.y - my_max) * my_isum);
    aP[10] = (__bf16)(__expf(sc.z - my_max) * my_isum);
    aP[11] = (__bf16)(__expf(sc.w - my_max) * my_isum);
    aP[12] = (__bf16)(__expf(sd.x - my_max) * my_isum);
    aP[13] = (__bf16)(__expf(sd.y - my_max) * my_isum);
    aP[14] = (__bf16)(__expf(sd.z - my_max) * my_isum);
    aP[15] = (__bf16)(__expf(sd.w - my_max) * my_isum);
    // B operand: lane L = dk column (L&15), V stored [dk][t] so keys loaded in table order
    const __bf16* vrow = Vb + (size_t)bh * DKK * TT + kc * 32;
    v16bf bV0 = load_a16x32(vrow, TT, lane);
    v16bf bV1 = load_a16x32(vrow + (size_t)16 * TT, TT, lane);
    o0 = wmma_bf16(aP, bV0, o0);
    o1 = wmma_bf16(aP, bV1, o1);
  }

  // Phase 4: write O as bf16 [B][T][H*DK] (heads re-concatenated for the out-proj GEMM)
#pragma unroll
  for (int r = 0; r < 8; ++r) {
    int row = qt * 16 + mtop + r;
    __bf16* op = Ob + ((size_t)(b_ * TT + row)) * DD + hd * DKK;
    op[nlo]      = (__bf16)o0[r];
    op[16 + nlo] = (__bf16)o1[r];
  }
}

// ---------------- out projection + bias + residual -> tmp (f32) ----------------
__global__ __launch_bounds__(32) void oproj_gemm_kernel(
    const __bf16* __restrict__ A, const __bf16* __restrict__ W, const float* __restrict__ bias,
    const float* __restrict__ resid, float* __restrict__ tmp) {
  const int lane = threadIdx.x;
  const int m0 = blockIdx.x * 16;
  const int n0 = blockIdx.y * 64;
  v8f acc0 = {}, acc1 = {}, acc2 = {}, acc3 = {};
  for (int kk = 0; kk < DD; kk += 32) {
    v16bf a = load_a16x32(A + (size_t)m0 * DD + kk, DD, lane);
    v16bf b0, b1, b2, b3;
#pragma unroll
    for (int s = 0; s < 16; ++s) {
      const int k = kk + ((lane >> 4) << 3) + (s & 7) + ((s >> 3) << 4);
      const __bf16* wp = W + (size_t)k * DD + n0 + (lane & 15);
      b0[s] = wp[0]; b1[s] = wp[16]; b2[s] = wp[32]; b3[s] = wp[48];
    }
    acc0 = wmma_bf16(a, b0, acc0);
    acc1 = wmma_bf16(a, b1, acc1);
    acc2 = wmma_bf16(a, b2, acc2);
    acc3 = wmma_bf16(a, b3, acc3);
  }
  const int mbase = (lane >> 4) << 3;
  const int nlo = lane & 15;
  v8f accs[4] = {acc0, acc1, acc2, acc3};
#pragma unroll
  for (int j = 0; j < 4; ++j) {
    int n = n0 + j * 16 + nlo;
    float bn = bias[n];
#pragma unroll
    for (int r = 0; r < 8; ++r) {
      int row = m0 + mbase + r;
      tmp[(size_t)row * DD + n] = accs[j][r] + bn + resid[(size_t)row * DD + n];
    }
  }
}

// ---------------- layernorm: one wave per row of 256 ----------------
__global__ __launch_bounds__(32) void layernorm_kernel(
    const float* __restrict__ tmp, const float* __restrict__ g, const float* __restrict__ bta,
    float* __restrict__ h, __bf16* __restrict__ hb) {
  int row = blockIdx.x;
  int lane = threadIdx.x;
  const float4* r4 = reinterpret_cast<const float4*>(tmp + (size_t)row * DD);
  float4 x0 = r4[lane];
  float4 x1 = r4[lane + 32];
  float s  = x0.x + x0.y + x0.z + x0.w + x1.x + x1.y + x1.z + x1.w;
  float sq = x0.x * x0.x + x0.y * x0.y + x0.z * x0.z + x0.w * x0.w
           + x1.x * x1.x + x1.y * x1.y + x1.z * x1.z + x1.w * x1.w;
#pragma unroll
  for (int off = 16; off > 0; off >>= 1) { s += __shfl_xor(s, off); sq += __shfl_xor(sq, off); }
  float mu = s * (1.0f / DD);
  float var = sq * (1.0f / DD) - mu * mu;
  float rs = rsqrtf(var + 1e-5f);
  float vals[8] = {x0.x, x0.y, x0.z, x0.w, x1.x, x1.y, x1.z, x1.w};
  int c0 = lane * 4, c1 = 128 + lane * 4;
  int cols[8] = {c0, c0 + 1, c0 + 2, c0 + 3, c1, c1 + 1, c1 + 2, c1 + 3};
#pragma unroll
  for (int e = 0; e < 8; ++e) {
    float y = (vals[e] - mu) * rs * g[cols[e]] + bta[cols[e]];
    h[(size_t)row * DD + cols[e]] = y;
    hb[(size_t)row * DD + cols[e]] = (__bf16)y;
  }
}

// ---------------- final head: out = h[:, -1, :] @ Wo + bo ----------------
__global__ __launch_bounds__(32) void head_kernel(
    const float* __restrict__ h, const float* __restrict__ Wo,
    const float* __restrict__ bo, float* __restrict__ out) {
  int i = threadIdx.x;
  if (i >= BB * 3) return;
  int b_ = i / 3, j = i % 3;
  const float* row = h + ((size_t)b_ * TT + (TT - 1)) * DD;
  float acc = bo[j];
  for (int n = 0; n < DD; ++n) acc += row[n] * Wo[n * 3 + j];
  out[i] = acc;
}

extern "C" void kernel_launch(void* const* d_in, const int* in_sizes, int n_in,
                              void* d_out, int out_size, void* d_ws, size_t ws_size,
                              hipStream_t stream) {
  (void)in_sizes; (void)n_in; (void)out_size; (void)ws_size;
  const float* x    = (const float*)d_in[0];
  const float* Wp   = (const float*)d_in[1];
  const float* bp   = (const float*)d_in[2];
  const float* pos  = (const float*)d_in[3];
  const float* Wqkv = (const float*)d_in[4];
  const float* bqkv = (const float*)d_in[5];
  const float* Wout = (const float*)d_in[6];
  const float* bout = (const float*)d_in[7];
  const float* ln_g = (const float*)d_in[8];
  const float* ln_b = (const float*)d_in[9];
  const float* Wo   = (const float*)d_in[10];
  const float* bo   = (const float*)d_in[11];
  float* out = (float*)d_out;

  // Workspace carve-up (~40 MB total)
  char* base = (char*)d_ws;
  size_t off = 0;
  auto take = [&](size_t bytes) -> void* {
    void* p = base + off;
    off += (bytes + 255) & ~(size_t)255;
    return p;
  };
  __bf16* wq_bf = (__bf16*)take((size_t)LLAYERS * DD * 3 * DD * sizeof(__bf16)); // 1.5 MB
  __bf16* wo_bf = (__bf16*)take((size_t)LLAYERS * DD * DD * sizeof(__bf16));     // 0.5 MB
  float*  h     = (float*)take((size_t)MTOT * DD * sizeof(float));               // 8 MB
  __bf16* hb    = (__bf16*)take((size_t)MTOT * DD * sizeof(__bf16));             // 4 MB
  __bf16* qb    = (__bf16*)take((size_t)BB * HH * TT * DKK * sizeof(__bf16));    // 4 MB
  __bf16* ktb   = (__bf16*)take((size_t)BB * HH * TT * DKK * sizeof(__bf16));    // 4 MB
  __bf16* vb    = (__bf16*)take((size_t)BB * HH * TT * DKK * sizeof(__bf16));    // 4 MB
  __bf16* ob    = (__bf16*)take((size_t)MTOT * DD * sizeof(__bf16));             // 4 MB
  float*  tmp   = (float*)take((size_t)MTOT * DD * sizeof(float));               // 8 MB

  {
    int n = LLAYERS * DD * 3 * DD;
    cvt_bf16_kernel<<<(n + 255) / 256, 256, 0, stream>>>(Wqkv, wq_bf, n);
  }
  {
    int n = LLAYERS * DD * DD;
    cvt_bf16_kernel<<<(n + 255) / 256, 256, 0, stream>>>(Wout, wo_bf, n);
  }

  inproj_kernel<<<MTOT, 256, 0, stream>>>(x, Wp, bp, pos, h, hb);

  for (int l = 0; l < LLAYERS; ++l) {
    qkv_gemm_kernel<<<dim3(MTOT / 16, (3 * DD) / 64), 32, 0, stream>>>(
        hb, wq_bf + (size_t)l * DD * 3 * DD, bqkv + (size_t)l * 3 * DD, qb, ktb, vb);
    attn_kernel<<<BB * HH * (TT / 16), 32, 0, stream>>>(qb, ktb, vb, ob);
    oproj_gemm_kernel<<<dim3(MTOT / 16, DD / 64), 32, 0, stream>>>(
        ob, wo_bf + (size_t)l * DD * DD, bout + (size_t)l * DD, h, tmp);
    layernorm_kernel<<<MTOT, 32, 0, stream>>>(tmp, ln_g + (size_t)l * DD, ln_b + (size_t)l * DD, h, hb);
  }

  head_kernel<<<1, 32, 0, stream>>>(h, Wo, bo, out);
}
